// PenguinVLVisionAttention_3650722202279
// MI455X (gfx1250) — matrix operand order repaired and checked by reading, not compiled, not run on hardware
//
#include <hip/hip_runtime.h>

// ---------------------------------------------------------------------------
// PenguinVLVisionAttention for MI455X (gfx1250, wave32, WMMA 16x16x32 bf16)
//
//   1) f32 -> bf16 conversion of hidden_states and Wq/Wk/Wv/Wo
//   2) WMMA GEMMs (async-to-LDS double buffered):
//        Q = X Wq^T (bf16), K = X Wk^T (bf16), V = X Wv^T (bf16, TRANSPOSED)
//   3) fused RMSNorm + 2-stream RoPE; Q pre-scaled by 1/sqrt(D)*log2(e)
//   4) flash attention per (head, segment, 128-row q block); 64-key blocks,
//      exp2 softmax, permlane16 row reductions, async double-buffered staging
//   5) WMMA GEMM: out = AO Wo^T (f32 -> d_out)
//
// Static shape: B=1, S=8192, HID=1024, H=16, HKV=8, D=128, 8 segs of 1024.
// ---------------------------------------------------------------------------

typedef unsigned short bf16_t;
typedef __attribute__((ext_vector_type(16))) __bf16 v16bf;
typedef __attribute__((ext_vector_type(8)))  __bf16 v8bf;
typedef __attribute__((ext_vector_type(8)))  float  v8f;
typedef int v4i __attribute__((vector_size(16)));   // matches builtin param type

#define S_TOK   8192
#define HID     1024
#define NH      16
#define NKV     8
#define HD      128
#define NSEG    8
#define SEGLEN  1024
#define QN      (NH * HD)    // 2048
#define KN      (NKV * HD)   // 1024

// ---------------- scalar helpers ----------------

__device__ __forceinline__ bf16_t f2bf(float f) {
    union { float f; unsigned u; } v; v.f = f;
    unsigned u = v.u;
    u += 0x7fffu + ((u >> 16) & 1u);      // round to nearest even
    return (bf16_t)(u >> 16);
}
__device__ __forceinline__ float bf2f(bf16_t h) {
    union { unsigned u; float f; } v; v.u = ((unsigned)h) << 16;
    return v.f;
}

#if __has_builtin(__builtin_amdgcn_exp2f)
__device__ __forceinline__ float fast_exp2(float x) { return __builtin_amdgcn_exp2f(x); }
#else
__device__ __forceinline__ float fast_exp2(float x) { return exp2f(x); }
#endif

// ---------------- async global -> LDS staging ----------------

#if __has_builtin(__builtin_amdgcn_global_load_async_to_lds_b128)
#define USE_ASYNC_LDS 1
#endif

__device__ __forceinline__ void cp16_lds(bf16_t* lds, const bf16_t* g) {
#if defined(USE_ASYNC_LDS)
    __builtin_amdgcn_global_load_async_to_lds_b128(
        (__attribute__((address_space(1))) v4i*)(void*)g,
        (__attribute__((address_space(3))) v4i*)(void*)lds, 0, 0);
#else
    *(uint4*)lds = *(const uint4*)g;
#endif
}
__device__ __forceinline__ void async_wait_all() {
#if defined(USE_ASYNC_LDS)
#if __has_builtin(__builtin_amdgcn_s_wait_asynccnt)
    __builtin_amdgcn_s_wait_asynccnt(0);
#else
    asm volatile("s_wait_asynccnt 0" ::: "memory");
#endif
#endif
}

// ---------------- 16-lane row reductions (VALU permlane, no LDS) ----------

#if __has_builtin(__builtin_amdgcn_permlane16)
__device__ __forceinline__ float pl16(float v, unsigned lo, unsigned hi) {
    return __uint_as_float(__builtin_amdgcn_permlane16(
        __float_as_uint(v), __float_as_uint(v), lo, hi, false, false));
}
__device__ __forceinline__ float red16_max(float v) {
    v = fmaxf(v, pl16(v, 0x67452301u, 0xEFCDAB89u));   // xor 1
    v = fmaxf(v, pl16(v, 0x54761032u, 0xDCFE98BAu));   // xor 2
    v = fmaxf(v, pl16(v, 0x32107654u, 0xBA98FEDCu));   // xor 4
    v = fmaxf(v, pl16(v, 0xFEDCBA98u, 0x76543210u));   // xor 8
    return v;
}
__device__ __forceinline__ float red16_sum(float v) {
    v += pl16(v, 0x67452301u, 0xEFCDAB89u);
    v += pl16(v, 0x54761032u, 0xDCFE98BAu);
    v += pl16(v, 0x32107654u, 0xBA98FEDCu);
    v += pl16(v, 0xFEDCBA98u, 0x76543210u);
    return v;
}
#else
__device__ __forceinline__ float red16_max(float v) {
#pragma unroll
    for (int m = 1; m < 16; m <<= 1) v = fmaxf(v, __shfl_xor(v, m, 32));
    return v;
}
__device__ __forceinline__ float red16_sum(float v) {
#pragma unroll
    for (int m = 1; m < 16; m <<= 1) v += __shfl_xor(v, m, 32);
    return v;
}
#endif

__device__ __forceinline__ float red32_sum(float v) {
#pragma unroll
    for (int m = 1; m < 32; m <<= 1) v += __shfl_xor(v, m, 32);
    return v;
}

// ---------------- WMMA fragment helpers ----------------

__device__ __forceinline__ v16bf ld_frag2(const bf16_t* p0, const bf16_t* p1) {
    v8bf lo = *(const v8bf*)p0;
    v8bf hi = *(const v8bf*)p1;
    v16bf r;
#pragma unroll
    for (int i = 0; i < 8; ++i) { r[i] = lo[i]; r[i + 8] = hi[i]; }
    return r;
}
__device__ __forceinline__ v16bf ld_frag16(const bf16_t* p) {
    return ld_frag2(p, p + 8);
}
__device__ __forceinline__ v8f wmma_bf16(v16bf a, v16bf b, v8f c) {
    return __builtin_amdgcn_wmma_f32_16x16x32_bf16(false, a, false, b,
                                                   (short)0, c, false, false);
}

// ---------------- 1) f32 -> bf16 conversion ----------------

__global__ __launch_bounds__(256) void k_f32_to_bf16(const float* __restrict__ src,
                                                     bf16_t* __restrict__ dst,
                                                     long n) {
    long i = ((long)blockIdx.x * 256 + threadIdx.x) * 4;
    if (i + 3 < n) {
        float4 v = *(const float4*)(src + i);
        dst[i + 0] = f2bf(v.x);
        dst[i + 1] = f2bf(v.y);
        dst[i + 2] = f2bf(v.z);
        dst[i + 3] = f2bf(v.w);
    } else {
        for (; i < n; ++i) dst[i] = f2bf(src[i]);
    }
}

// ---------------- 2/5) tiled bf16 WMMA GEMM ----------------
// C[M,N] = A[M,K] * W[N,K]^T.  Block tile 128x128; 8 waves in 4(M)x2(N);
// each wave 32x64 = 8 wmma per 32-wide K step.  Double-buffered LDS filled
// with async global->LDS copies; rows padded to 40 for bank-spread B128 reads.
// MODE: 0 = bf16 row-major out, 1 = f32 row-major out,
//       2 = bf16 transposed out (Cout[col*S_TOK + row], packed b128 stores)

template <int MODE>
__global__ __launch_bounds__(256) void k_gemm_bf16(const bf16_t* __restrict__ A,
                                                   const bf16_t* __restrict__ W,
                                                   void* __restrict__ Cout,
                                                   int M, int N, int K) {
    __shared__ alignas(16) bf16_t AL[2][128 * 40];
    __shared__ alignas(16) bf16_t WL[2][128 * 40];

    const int tid  = threadIdx.x;
    const int wid  = tid >> 5;
    const int lane = tid & 31;
    const int l16  = lane & 15;
    const int hi   = lane >> 4;
    const int wm   = wid >> 1;             // 0..3 -> 32-row strip
    const int wn   = wid & 1;              // 0..1 -> 64-col strip
    const int blockM = blockIdx.y * 128;
    const int blockN = blockIdx.x * 128;

    // staging assignment: each thread copies 32B of A and 32B of W per step
    const int sr = tid >> 1;               // 0..127
    const int sc = (tid & 1) * 16;         // 0 / 16

    v8f acc[8];
#pragma unroll
    for (int f = 0; f < 8; ++f)
#pragma unroll
        for (int e = 0; e < 8; ++e) acc[f][e] = 0.f;

    // prologue: stage k0 = 0 into buffer 0
    {
        const bf16_t* ga = A + (size_t)(blockM + sr) * K + sc;
        const bf16_t* gw = W + (size_t)(blockN + sr) * K + sc;
        cp16_lds(&AL[0][sr * 40 + sc], ga);
        cp16_lds(&AL[0][sr * 40 + sc + 8], ga + 8);
        cp16_lds(&WL[0][sr * 40 + sc], gw);
        cp16_lds(&WL[0][sr * 40 + sc + 8], gw + 8);
    }
    async_wait_all();
    __syncthreads();

    int cur = 0;
    const int o = hi * 8;
    for (int k0 = 0; k0 < K; k0 += 32) {
        if (k0 + 32 < K) {   // stage next tile into the other buffer
            const bf16_t* ga = A + (size_t)(blockM + sr) * K + (k0 + 32) + sc;
            const bf16_t* gw = W + (size_t)(blockN + sr) * K + (k0 + 32) + sc;
            cp16_lds(&AL[cur ^ 1][sr * 40 + sc], ga);
            cp16_lds(&AL[cur ^ 1][sr * 40 + sc + 8], ga + 8);
            cp16_lds(&WL[cur ^ 1][sr * 40 + sc], gw);
            cp16_lds(&WL[cur ^ 1][sr * 40 + sc + 8], gw + 8);
        }

        const bf16_t* ar0 = &AL[cur][(wm * 32 + l16) * 40];
        const bf16_t* ar1 = ar0 + 16 * 40;
        v16bf a0 = ld_frag2(ar0 + o, ar0 + o + 16);
        v16bf a1 = ld_frag2(ar1 + o, ar1 + o + 16);
#pragma unroll
        for (int nt = 0; nt < 4; ++nt) {
            v16bf b = ld_frag2(&WL[cur][(wn * 64 + nt * 16 + l16) * 40 + o], &WL[cur][(wn * 64 + nt * 16 + l16) * 40 + o + 16]);
            acc[nt]     = wmma_bf16(a0, b, acc[nt]);
            acc[4 + nt] = wmma_bf16(a1, b, acc[4 + nt]);
        }

        async_wait_all();
        __syncthreads();
        cur ^= 1;
    }

    // epilogue; C layout: VGPR g -> row g + 8*hi (8 consecutive rows per lane)
#pragma unroll
    for (int f = 0; f < 8; ++f) {
        const int j  = f >> 2;             // which 16-row subtile
        const int nt = f & 3;
        const int col  = blockN + wn * 64 + nt * 16 + l16;
        const int rowb = blockM + wm * 32 + j * 16 + hi * 8;
        if (MODE == 2) {
            // transposed bf16: 8 consecutive rows at fixed col -> one b128
            unsigned u0 = (unsigned)f2bf(acc[f][0]) | ((unsigned)f2bf(acc[f][1]) << 16);
            unsigned u1 = (unsigned)f2bf(acc[f][2]) | ((unsigned)f2bf(acc[f][3]) << 16);
            unsigned u2 = (unsigned)f2bf(acc[f][4]) | ((unsigned)f2bf(acc[f][5]) << 16);
            unsigned u3 = (unsigned)f2bf(acc[f][6]) | ((unsigned)f2bf(acc[f][7]) << 16);
            uint4 pk; pk.x = u0; pk.y = u1; pk.z = u2; pk.w = u3;
            *(uint4*)((bf16_t*)Cout + (size_t)col * S_TOK + rowb) = pk;
        } else {
#pragma unroll
            for (int g = 0; g < 8; ++g) {
                float v = acc[f][g];
                size_t idx = (size_t)(rowb + g) * N + col;
                if (MODE == 1) ((float*)Cout)[idx]  = v;
                else           ((bf16_t*)Cout)[idx] = f2bf(v);
            }
        }
    }
}

// ---------------- 3) fused RMSNorm + multimodal RoPE (in place) ------------
// One wave per (token, head) row.  Lane holds d = {lane, lane+32, lane+64,
// lane+96}: rotate-half partners are in-lane.  Q rows additionally scaled by
// 1/sqrt(D) * log2(e) so attention softmax can run in pure exp2.

__global__ __launch_bounds__(256) void k_normrope(bf16_t* __restrict__ Qb,
                                                  bf16_t* __restrict__ Kb,
                                                  const float* __restrict__ qw,
                                                  const float* __restrict__ kw,
                                                  const float* __restrict__ cosT,
                                                  const float* __restrict__ sinT) {
    const int wid  = threadIdx.x >> 5;
    const int lane = threadIdx.x & 31;
    const long rowid = (long)blockIdx.x * 8 + wid;   // S*24 rows total
    const int token = (int)(rowid / (NH + NKV));
    const int slot  = (int)(rowid % (NH + NKV));

    bf16_t* row;
    const float* w;
    float oscale;
    const float QSC = 0.08838834764831845f * 1.44269504088896340f;
    if (slot < NH) { row = Qb + (size_t)token * QN + slot * HD;        w = qw; oscale = QSC; }
    else           { row = Kb + (size_t)token * KN + (slot - NH) * HD; w = kw; oscale = 1.f; }

    float x[4];
#pragma unroll
    for (int j = 0; j < 4; ++j) x[j] = bf2f(row[lane + 32 * j]);

    float ss = x[0]*x[0] + x[1]*x[1] + x[2]*x[2] + x[3]*x[3];
    ss = red32_sum(ss);
    const float inv = rsqrtf(ss * (1.f / HD) + 1e-6f);

    float y[4], c[4], s[4];
#pragma unroll
    for (int j = 0; j < 4; ++j) {
        int d = lane + 32 * j;
        y[j] = x[j] * inv * w[d];
        int st = (d < 64) ? 0 : 1;            // multimodal stream select
        long ti = (long)st * S_TOK * HD + (long)token * HD + d;
        c[j] = cosT[ti];
        s[j] = sinT[ti];
    }
    float r0 = (y[0] * c[0] - y[2] * s[0]) * oscale;
    float r1 = (y[1] * c[1] - y[3] * s[1]) * oscale;
    float r2 = (y[2] * c[2] + y[0] * s[2]) * oscale;
    float r3 = (y[3] * c[3] + y[1] * s[3]) * oscale;

    row[lane +  0] = f2bf(r0);
    row[lane + 32] = f2bf(r1);
    row[lane + 64] = f2bf(r2);
    row[lane + 96] = f2bf(r3);
}

// ---------------- 4) flash attention per (head, segment, q-block) ----------
// 256 threads = 8 waves, each wave owns 16 q rows.  64-key blocks: one
// softmax reduction chain + one accumulator rescale covers 4 score tiles,
// halving the per-key VALU overhead vs 32-key blocks.  K staged row-major
// (stride 136) and V from the pre-transposed Vt[KN][S] buffer (stride 72),
// both via double-buffered async global->LDS copies.  P round-trips through
// wave-private LDS (per-wave in-order LDS => compiler fence only).

__global__ __launch_bounds__(256) void k_attn(const bf16_t* __restrict__ Qb,
                                              const bf16_t* __restrict__ Kb,
                                              const bf16_t* __restrict__ Vt,
                                              bf16_t* __restrict__ AO) {
    __shared__ alignas(16) bf16_t KL[2][64 * 136];   // 64 keys x 128 d
    __shared__ alignas(16) bf16_t VL[2][128 * 72];   // 128 d x 64 keys
    __shared__ alignas(16) bf16_t PLDS[8 * 16 * 72]; // per-wave 16 x 64 probs

    const int h    = blockIdx.x;            // 0..15
    const int seg  = blockIdx.y;            // 0..7
    const int qb   = blockIdx.z;            // 0..7
    const int kvh  = h >> 1;                // GQA groups = 2
    const int tid  = threadIdx.x;
    const int wid  = tid >> 5;
    const int lane = tid & 31;
    const int l16  = lane & 15;
    const int hi   = lane >> 4;
    const int o    = hi * 8;

    const int qtok0 = seg * SEGLEN + qb * 128 + wid * 16;

    // K staging: thread copies 64B of one key row
    const int kr = tid >> 2, kc = (tid & 3) * 32;
    const bf16_t* gkb = Kb + (size_t)(seg * SEGLEN + kr) * KN + kvh * HD + kc;
    // Vt staging: thread copies 64B of one d row (keys contiguous)
    const int vr = tid >> 1, vc = (tid & 1) * 32;
    const bf16_t* gvb = Vt + (size_t)(kvh * HD + vr) * S_TOK + seg * SEGLEN + vc;

    // Q fragments (pre-scaled at normrope)
    const bf16_t* qrow = Qb + (size_t)(qtok0 + l16) * QN + h * HD;
    v16bf qf[4];
#pragma unroll
    for (int kk = 0; kk < 4; ++kk)
        qf[kk] = ld_frag2(qrow + kk * 32 + o, qrow + kk * 32 + o + 16);

    float mrow[8], lrow[8];
    v8f oacc[8];
#pragma unroll
    for (int g = 0; g < 8; ++g) { mrow[g] = -1e30f; lrow[g] = 0.f; }
#pragma unroll
    for (int nt = 0; nt < 8; ++nt)
#pragma unroll
        for (int e = 0; e < 8; ++e) oacc[nt][e] = 0.f;

    // prologue: stage key block 0 into buffer 0
#pragma unroll
    for (int q4 = 0; q4 < 4; ++q4) {
        cp16_lds(&KL[0][kr * 136 + kc + q4 * 8], gkb + q4 * 8);
        cp16_lds(&VL[0][vr * 72 + vc + q4 * 8],  gvb + q4 * 8);
    }
    async_wait_all();
    __syncthreads();

    int cur = 0;
    const int NB = SEGLEN / 64;
    for (int kb = 0; kb < NB; ++kb) {
        if (kb + 1 < NB) {               // stage next block into other buffer
            const bf16_t* gk = gkb + (size_t)(kb + 1) * 64 * KN;
            const bf16_t* gv = gvb + (size_t)(kb + 1) * 64;
#pragma unroll
            for (int q4 = 0; q4 < 4; ++q4) {
                cp16_lds(&KL[cur ^ 1][kr * 136 + kc + q4 * 8], gk + q4 * 8);
                cp16_lds(&VL[cur ^ 1][vr * 72 + vc + q4 * 8],  gv + q4 * 8);
            }
        }

        // S = Q K^T over four 16-key tiles (already in log2 scale)
        v8f sc4[4];
#pragma unroll
        for (int j = 0; j < 4; ++j) {
#pragma unroll
            for (int e = 0; e < 8; ++e) sc4[j][e] = 0.f;
#pragma unroll
            for (int kk = 0; kk < 4; ++kk) {
                v16bf bk = ld_frag2(&KL[cur][(j * 16 + l16) * 136 + kk * 32 + o], &KL[cur][(j * 16 + l16) * 136 + kk * 32 + o + 16]);
                sc4[j] = wmma_bf16(qf[kk], bk, sc4[j]);
            }
        }

        // online softmax (exp2 domain): one reduction chain over 64 keys
        float fac[8];
#pragma unroll
        for (int g = 0; g < 8; ++g) {
            float a0 = sc4[0][g], a1 = sc4[1][g];
            float a2 = sc4[2][g], a3 = sc4[3][g];
            float mx = red16_max(fmaxf(fmaxf(a0, a1), fmaxf(a2, a3)));
            float nm = fmaxf(mrow[g], mx);
            float p0 = fast_exp2(a0 - nm);
            float p1 = fast_exp2(a1 - nm);
            float p2 = fast_exp2(a2 - nm);
            float p3 = fast_exp2(a3 - nm);
            float rs = red16_sum((p0 + p1) + (p2 + p3));
            float f  = fast_exp2(mrow[g] - nm);
            mrow[g] = nm;
            lrow[g] = lrow[g] * f + rs;
            fac[g]  = f;
            int prow = wid * 1152 + (g + hi * 8) * 72;
            PLDS[prow + l16]      = f2bf(p0);
            PLDS[prow + 16 + l16] = f2bf(p1);
            PLDS[prow + 32 + l16] = f2bf(p2);
            PLDS[prow + 48 + l16] = f2bf(p3);
        }
#pragma unroll
        for (int nt = 0; nt < 8; ++nt)
#pragma unroll
            for (int g = 0; g < 8; ++g) oacc[nt][g] *= fac[g];

        // wave-private P staging: hardware keeps per-wave LDS ops in order;
        // only block compiler reordering here.
        asm volatile("" ::: "memory");

        const bf16_t* prp = &PLDS[wid * 1152 + l16 * 72];
        v16bf pf0 = ld_frag2(prp + o,      prp + o + 16);        // keys  0..31
        v16bf pf1 = ld_frag2(prp + 32 + o, prp + 32 + o + 16);   // keys 32..63
#pragma unroll
        for (int nt = 0; nt < 8; ++nt) {
            const bf16_t* vrow = &VL[cur][(nt * 16 + l16) * 72];
            v16bf bv0 = ld_frag2(vrow + o, vrow + o + 16);
            oacc[nt] = wmma_bf16(pf0, bv0, oacc[nt]);
            v16bf bv1 = ld_frag2(vrow + 32 + o, vrow + 32 + o + 16);
            oacc[nt] = wmma_bf16(pf1, bv1, oacc[nt]);
        }

        async_wait_all();
        __syncthreads();
        cur ^= 1;
    }

    // normalize and write AO[token][h*128 + d]
#pragma unroll
    for (int nt = 0; nt < 8; ++nt)
#pragma unroll
        for (int g = 0; g < 8; ++g) {
            int row = qtok0 + g + hi * 8;
            int col = h * HD + nt * 16 + l16;
            AO[(size_t)row * QN + col] = f2bf(oacc[nt][g] / lrow[g]);
        }
}

// ---------------- host side ----------------

static inline size_t align256(size_t x) { return (x + 255) & ~(size_t)255; }

extern "C" void kernel_launch(void* const* d_in, const int* in_sizes, int n_in,
                              void* d_out, int out_size, void* d_ws, size_t ws_size,
                              hipStream_t stream) {
    (void)in_sizes; (void)n_in; (void)out_size; (void)ws_size;

    const float* hidden = (const float*)d_in[0];
    /* d_in[1] = cu_seqlens: equal-length static segments, handled by shape */
    const float* cosT = (const float*)d_in[2];
    const float* sinT = (const float*)d_in[3];
    const float* Wq   = (const float*)d_in[4];
    const float* Wk   = (const float*)d_in[5];
    const float* Wv   = (const float*)d_in[6];
    const float* Wo   = (const float*)d_in[7];
    const float* qw   = (const float*)d_in[8];
    const float* kw   = (const float*)d_in[9];

    char* ws = (char*)d_ws;
    size_t off = 0;
    bf16_t* hid_bf = (bf16_t*)(ws + off); off = align256(off + (size_t)S_TOK * HID * 2);
    bf16_t* wq_bf  = (bf16_t*)(ws + off); off = align256(off + (size_t)QN * HID * 2);
    bf16_t* wk_bf  = (bf16_t*)(ws + off); off = align256(off + (size_t)KN * HID * 2);
    bf16_t* wv_bf  = (bf16_t*)(ws + off); off = align256(off + (size_t)KN * HID * 2);
    bf16_t* wo_bf  = (bf16_t*)(ws + off); off = align256(off + (size_t)HID * QN * 2);
    bf16_t* qbuf   = (bf16_t*)(ws + off); off = align256(off + (size_t)S_TOK * QN * 2);
    bf16_t* kbuf   = (bf16_t*)(ws + off); off = align256(off + (size_t)S_TOK * KN * 2);
    bf16_t* vtbuf  = (bf16_t*)(ws + off); off = align256(off + (size_t)KN * S_TOK * 2);
    bf16_t* aobuf  = (bf16_t*)(ws + off); off = align256(off + (size_t)S_TOK * QN * 2);

    // 1) conversions
    {
        struct { const float* s; bf16_t* d; long n; } cv[5] = {
            { hidden, hid_bf, (long)S_TOK * HID },
            { Wq,     wq_bf,  (long)QN * HID },
            { Wk,     wk_bf,  (long)KN * HID },
            { Wv,     wv_bf,  (long)KN * HID },
            { Wo,     wo_bf,  (long)HID * QN },
        };
        for (int i = 0; i < 5; ++i) {
            long t = (cv[i].n + 3) / 4;
            long grid = (t + 255) / 256;
            k_f32_to_bf16<<<(unsigned)grid, 256, 0, stream>>>(cv[i].s, cv[i].d, cv[i].n);
        }
    }

    // 2) projections (V written pre-transposed)
    k_gemm_bf16<0><<<dim3(QN / 128, S_TOK / 128), 256, 0, stream>>>(
        hid_bf, wq_bf, qbuf, S_TOK, QN, HID);
    k_gemm_bf16<0><<<dim3(KN / 128, S_TOK / 128), 256, 0, stream>>>(
        hid_bf, wk_bf, kbuf, S_TOK, KN, HID);
    k_gemm_bf16<2><<<dim3(KN / 128, S_TOK / 128), 256, 0, stream>>>(
        hid_bf, wv_bf, vtbuf, S_TOK, KN, HID);

    // 3) RMSNorm + RoPE on Q and K (in place; Q pre-scaled for exp2 softmax)
    k_normrope<<<(S_TOK * (NH + NKV)) / 8, 256, 0, stream>>>(
        qbuf, kbuf, qw, kw, cosT, sinT);

    // 4) block-diagonal GQA flash attention (64-key blocks)
    k_attn<<<dim3(NH, NSEG, SEGLEN / 128), 256, 0, stream>>>(
        qbuf, kbuf, vtbuf, aobuf);

    // 5) output projection -> fp32 d_out
    k_gemm_bf16<1><<<dim3(HID / 128, S_TOK / 128), 256, 0, stream>>>(
        aobuf, wo_bf, d_out, S_TOK, HID, QN);
}
